// EGNNEncoder_61916248539246
// MI455X (gfx1250) — matrix operand order repaired and checked by reading, not compiled, not run on hardware
//
#include <hip/hip_runtime.h>
#include <hip/hip_fp16.h>

// EGNN encoder for MI455X (gfx1250, wave32, WMMA).
// GEMMs via v_wmma_f32_16x16x32_f16. A fragments for gather-GEMMs are built
// directly from global memory (2x contiguous 8-float runs per lane -> float4
// loads). B weight panels and inter-GEMM activations live in LDS in
// fragment-major layout so every WMMA operand read is a contiguous 32B
// (2x ds_load_b128). Segment sums via native f32 global atomics; the whole
// ~27MB working set is resident in the 192MB L2.

typedef __attribute__((ext_vector_type(16))) _Float16 v16h;
typedef __attribute__((ext_vector_type(8)))  float    v8f;

#define N_NODES 50000
#define N_EDGES 800000
#define HID 64
#define N_LAYERS 8

__device__ __forceinline__ float sigm_(float z) { return 1.0f / (1.0f + __expf(-z)); }
__device__ __forceinline__ float silu_(float z) { return z * sigm_(z); }

// Wave-local LDS ordering: DS ops are in-order within a wave (ISA 7.3); this is
// a compiler memory fence plus an explicit CDNA5 dscnt wait.
__device__ __forceinline__ void lds_fence() {
  asm volatile("s_wait_dscnt 0x0" ::: "memory");
}

__device__ __forceinline__ v8f vzero8() {
  v8f z;
#pragma unroll
  for (int e = 0; e < 8; ++e) z[e] = 0.0f;
  return z;
}

__device__ __forceinline__ v8f wmma_(v16h a, v16h b, v8f c) {
  return __builtin_amdgcn_wmma_f32_16x16x32_f16(false, a, false, b, (short)0, c, false, false);
}

// A-fragment element (row m, k-offset koff within a 32-wide chunk) lives at
// lane = m + 16*hi, element e  (ISA 7.12.2 16-bit A layout):
//   koff = (e&7) + 16*(e>=8) + 8*hi
__device__ __forceinline__ int a_frag_idx(int m, int koff) {
  const int hi = (koff >> 3) & 1;
  const int e  = (koff & 7) | ((koff >> 4) << 3);
  return ((m + (hi << 4)) << 4) + e;  // lane*16 + e
}

// Build an A fragment straight from a global f32 row: elements e=0..7 come from
// rowp[kb..kb+7], e=8..15 from rowp[kb+16..kb+23]  (kb = chunk*32 + 8*hi).
__device__ __forceinline__ v16h afrag_global(const float* rowp, int kb) {
  const float4 p0 = *(const float4*)(rowp + kb);
  const float4 p1 = *(const float4*)(rowp + kb + 4);
  const float4 p2 = *(const float4*)(rowp + kb + 16);
  const float4 p3 = *(const float4*)(rowp + kb + 20);
  v16h a;
  a[0] = (_Float16)p0.x; a[1] = (_Float16)p0.y; a[2]  = (_Float16)p0.z; a[3]  = (_Float16)p0.w;
  a[4] = (_Float16)p1.x; a[5] = (_Float16)p1.y; a[6]  = (_Float16)p1.z; a[7]  = (_Float16)p1.w;
  a[8] = (_Float16)p2.x; a[9] = (_Float16)p2.y; a[10] = (_Float16)p2.z; a[11] = (_Float16)p2.w;
  a[12] = (_Float16)p3.x; a[13] = (_Float16)p3.y; a[14] = (_Float16)p3.z; a[15] = (_Float16)p3.w;
  return a;
}

// One K=32 chunk against a fragment-major B panel: 4 N-tiles of WMMA.
// Panel layout: [(chunk*4 + t)*512 + lane*16 + e]  -> contiguous 32B per lane.
__device__ __forceinline__ void gemm_chunk(v16h a, const _Float16* BsF, int chunk,
                                           v8f acc[4], int lane) {
#pragma unroll
  for (int t = 0; t < 4; ++t) {
    v16h b = *(const v16h*)(BsF + ((((chunk << 2) + t) << 9) + (lane << 4)));
    acc[t] = wmma_(a, b, acc[t]);
  }
}

// Stage a (K x 64) weight matrix (row-major w[n*ldw + k], n = out index) into a
// fragment-major B panel, zero-padding K up to kPad (multiple of 32).
// B layout (ISA 7.12.2): lane = nlo + 16*((koff>>3)&1), e = (koff&7) + 8*(koff>=16).
__device__ __forceinline__ void stage_bpanel(_Float16* dst, const float* w, int ldw,
                                             int kReal, int kPad, int tid, int nthr) {
  for (int i = tid; i < kPad * HID; i += nthr) {
    const int k = i >> 6, n = i & 63;
    const int chunk = k >> 5, koff = k & 31;
    const int t = n >> 4, nlo = n & 15;
    const int lane = nlo + (((koff >> 3) & 1) << 4);
    const int e = (koff & 7) | ((koff >> 4) << 3);
    dst[(((chunk << 2) + t) << 9) + (lane << 4) + e] =
        (_Float16)((k < kReal) ? w[n * ldw + k] : 0.0f);
  }
}

// ---------------------------------------------------------------------------
// Edge kernel: one wave <-> one 16-edge tile.
// ---------------------------------------------------------------------------
#define EW 4  // waves per block
__global__ __launch_bounds__(32 * EW)
void egnn_edge_kernel(const float* __restrict__ h, const float* __restrict__ xw,
                      const long long* __restrict__ ei,
                      const float* __restrict__ ew1, const float* __restrict__ eb1,
                      const float* __restrict__ ew2, const float* __restrict__ eb2,
                      const float* __restrict__ attw, const float* __restrict__ attb,
                      const float* __restrict__ cw1, const float* __restrict__ cb1,
                      const float* __restrict__ cw2,
                      float* __restrict__ agg, float* __restrict__ cagg,
                      float* __restrict__ cnt, int layer, int nTiles) {
  __shared__ alignas(32) _Float16 sW1F[5 * 4 * 512];   // edge_w1^T, K padded 130->160
  __shared__ alignas(32) _Float16 sW2F[2 * 4 * 512];   // edge_w2^T
  __shared__ alignas(32) _Float16 sCW1F[2 * 4 * 512];  // coord_w1^T
  __shared__ float sB1[HID], sB2[HID], sAtt[HID], sCB1[HID], sCW2v[HID];
  __shared__ float sAttB;
  __shared__ alignas(32) _Float16 sMF[EW][2 * 512];    // fragment-major activations
  __shared__ alignas(32) _Float16 sFh[EW][16 * HID];   // row-major f16 (dot products)
  __shared__ int   sRow[EW][16];
  __shared__ int   sCol[EW][16];
  __shared__ float sRad[EW][16];
  __shared__ float sCd[EW][32];
  __shared__ float sS[EW][16];

  const int tid = threadIdx.x;
  stage_bpanel(sW1F, ew1 + layer * HID * 130, 130, 130, 160, tid, blockDim.x);
  stage_bpanel(sW2F, ew2 + layer * HID * HID, HID, HID, HID, tid, blockDim.x);
  stage_bpanel(sCW1F, cw1 + layer * HID * HID, HID, HID, HID, tid, blockDim.x);
  for (int i = tid; i < HID; i += blockDim.x) {
    sB1[i]   = eb1[layer * HID + i];
    sB2[i]   = eb2[layer * HID + i];
    sAtt[i]  = attw[layer * HID + i];
    sCB1[i]  = cb1[layer * HID + i];
    sCW2v[i] = cw2[layer * HID + i];
  }
  if (tid == 0) sAttB = attb[layer];
  __syncthreads();

  const int w = tid >> 5;
  const int lane = tid & 31;
  const int m = lane & 15;
  const int hi = lane >> 4;
  const int nlo = lane & 15;
  const int wstride = gridDim.x * EW;

  for (int tile = blockIdx.x * EW + w; tile < nTiles; tile += wstride) {
    // 1. per-edge metadata
    if (lane < 16) {
      const int e = lane;
      const long long g = (long long)tile * 16 + e;
      const int r = (int)ei[g];
      const int c = (int)ei[(long long)N_EDGES + g];
      sRow[w][e] = r;
      sCol[w][e] = c;
      const float dx = xw[2 * r] - xw[2 * c];
      const float dy = xw[2 * r + 1] - xw[2 * c + 1];
      sCd[w][2 * e] = dx;
      sCd[w][2 * e + 1] = dy;
      sRad[w][e] = dx * dx + dy * dy;
      const long long g2 = g + (long long)wstride * 16;
      if (g2 < N_EDGES) __builtin_prefetch(&ei[g2], 0, 0);  // global_prefetch_b8
    }
    lds_fence();
    const int r0 = sRow[w][m];
    const int c0 = sCol[w][m];
    const float rad = sRad[w][m];

    // 2. GEMM1: edge_in(16x160) * W1^T(160x64); A chunks straight from global h
    v8f acc[4];
#pragma unroll
    for (int t = 0; t < 4; ++t) acc[t] = vzero8();
    {
      const float* hr = h + (long long)r0 * HID;
      const float* hc = h + (long long)c0 * HID;
      const int kb = hi << 3;
      gemm_chunk(afrag_global(hr, kb), sW1F, 0, acc, lane);
      gemm_chunk(afrag_global(hr, 32 + kb), sW1F, 1, acc, lane);
      gemm_chunk(afrag_global(hc, kb), sW1F, 2, acc, lane);
      gemm_chunk(afrag_global(hc, 32 + kb), sW1F, 3, acc, lane);
      v16h a4;  // tail chunk k=128..159: [radial, 1, 0...] -> only hi==0 lanes
#pragma unroll
      for (int e = 0; e < 16; ++e) a4[e] = (_Float16)0.0f;
      a4[0] = hi ? (_Float16)0.0f : (_Float16)rad;
      a4[1] = hi ? (_Float16)0.0f : (_Float16)1.0f;
      gemm_chunk(a4, sW1F, 4, acc, lane);
    }
    // 3. bias + SiLU -> fragment-major f16 stage
#pragma unroll
    for (int t = 0; t < 4; ++t)
#pragma unroll
      for (int r = 0; r < 8; ++r) {
        const int M = r + (hi << 3), N = t * 16 + nlo;
        sMF[w][((N >> 5) << 9) + a_frag_idx(M, N & 31)] = (_Float16)silu_(acc[t][r] + sB1[N]);
      }
    lds_fence();
    // 4. GEMM2: (16x64)*(64x64)
#pragma unroll
    for (int t = 0; t < 4; ++t) acc[t] = vzero8();
#pragma unroll
    for (int c = 0; c < 2; ++c) {
      v16h a = *(const v16h*)(&sMF[w][(c << 9) + (lane << 4)]);
      gemm_chunk(a, sW2F, c, acc, lane);
    }
    // 5. bias + SiLU: keep in registers, publish rows for the gate dot
    float vals[4][8];
#pragma unroll
    for (int t = 0; t < 4; ++t)
#pragma unroll
      for (int r = 0; r < 8; ++r) {
        const int M = r + (hi << 3), N = t * 16 + nlo;
        vals[t][r] = silu_(acc[t][r] + sB2[N]);
        sFh[w][M * HID + N] = (_Float16)vals[t][r];
      }
    lds_fence();
    // 6. attention gate per edge (wide v16h row reads)
    if (lane < 16) {
      const _Float16* rowp = &sFh[w][lane * HID];
      float s = sAttB;
#pragma unroll
      for (int q = 0; q < 4; ++q) {
        v16h vv = *(const v16h*)(rowp + q * 16);
#pragma unroll
        for (int e = 0; e < 16; ++e) s += (float)vv[e] * sAtt[q * 16 + e];
      }
      sS[w][lane] = sigm_(s);
    }
    lds_fence();
    // 7. gate from registers: scatter gated messages + restage as coord-A frags
    {
      float gM[8];
      int rowM[8];
#pragma unroll
      for (int r = 0; r < 8; ++r) {
        const int M = r + (hi << 3);
        gM[r] = sS[w][M];
        rowM[r] = sRow[w][M];
      }
#pragma unroll
      for (int t = 0; t < 4; ++t)
#pragma unroll
        for (int r = 0; r < 8; ++r) {
          const int M = r + (hi << 3), N = t * 16 + nlo;
          const float gv = vals[t][r] * gM[r];
          unsafeAtomicAdd(&agg[(long long)rowM[r] * HID + N], gv);
          sMF[w][((N >> 5) << 9) + a_frag_idx(M, N & 31)] = (_Float16)gv;
        }
    }
    lds_fence();
    // 8. coord MLP layer1: (16x64)*(64x64), bias+SiLU -> row-major for cu dot
#pragma unroll
    for (int t = 0; t < 4; ++t) acc[t] = vzero8();
#pragma unroll
    for (int c = 0; c < 2; ++c) {
      v16h a = *(const v16h*)(&sMF[w][(c << 9) + (lane << 4)]);
      gemm_chunk(a, sCW1F, c, acc, lane);
    }
#pragma unroll
    for (int t = 0; t < 4; ++t)
#pragma unroll
      for (int r = 0; r < 8; ++r) {
        const int M = r + (hi << 3), N = t * 16 + nlo;
        sFh[w][M * HID + N] = (_Float16)silu_(acc[t][r] + sCB1[N]);
      }
    lds_fence();
    // 9. cu = c1 . coord_w2 per edge
    if (lane < 16) {
      const _Float16* rowp = &sFh[w][lane * HID];
      float s = 0.0f;
#pragma unroll
      for (int q = 0; q < 4; ++q) {
        v16h vv = *(const v16h*)(rowp + q * 16);
#pragma unroll
        for (int e = 0; e < 16; ++e) s += (float)vv[e] * sCW2v[q * 16 + e];
      }
      sS[w][lane] = s;
    }
    lds_fence();
    // 10. coordinate scatter + counts
    if (lane < 16) {
      const int e = lane, r = sRow[w][e];
      const float cu = sS[w][e];
      unsafeAtomicAdd(&cagg[2 * r],     sCd[w][2 * e] * cu);
      unsafeAtomicAdd(&cagg[2 * r + 1], sCd[w][2 * e + 1] * cu);
      unsafeAtomicAdd(&cnt[r], 1.0f);
    }
  }
}

// ---------------------------------------------------------------------------
// Node kernel: one wave <-> one 16-node tile; A = [h | agg] from global.
// ---------------------------------------------------------------------------
#define NWV 8
__global__ __launch_bounds__(32 * NWV)
void egnn_node_kernel(float* __restrict__ h, float* __restrict__ xw,
                      const float* __restrict__ agg, const float* __restrict__ cagg,
                      const float* __restrict__ cnt,
                      const float* __restrict__ nw1, const float* __restrict__ nb1,
                      const float* __restrict__ nw2, const float* __restrict__ nb2,
                      int layer, int nTiles) {
  __shared__ alignas(32) _Float16 sW1F[4 * 4 * 512];  // node_w1^T (128x64)
  __shared__ alignas(32) _Float16 sW2F[2 * 4 * 512];  // node_w2^T (64x64)
  __shared__ float sB1[HID], sB2[HID];
  __shared__ alignas(32) _Float16 sHF[NWV][2 * 512];

  const int tid = threadIdx.x;
  stage_bpanel(sW1F, nw1 + layer * HID * 128, 128, 128, 128, tid, blockDim.x);
  stage_bpanel(sW2F, nw2 + layer * HID * HID, HID, HID, HID, tid, blockDim.x);
  for (int i = tid; i < HID; i += blockDim.x) {
    sB1[i] = nb1[layer * HID + i];
    sB2[i] = nb2[layer * HID + i];
  }
  __syncthreads();

  const int w = tid >> 5, lane = tid & 31;
  const int m = lane & 15, hi = lane >> 4, nlo = lane & 15;
  const int wstride = gridDim.x * NWV;

  for (int tile = blockIdx.x * NWV + w; tile < nTiles; tile += wstride) {
    const int nb = tile * 16;
    const long long node0 = nb + m;
    const float* hp = h + node0 * HID;
    const float* ap = agg + node0 * HID;
    const int kb = hi << 3;

    v8f acc[4];
#pragma unroll
    for (int t = 0; t < 4; ++t) acc[t] = vzero8();
    gemm_chunk(afrag_global(hp, kb), sW1F, 0, acc, lane);
    gemm_chunk(afrag_global(hp, 32 + kb), sW1F, 1, acc, lane);
    gemm_chunk(afrag_global(ap, kb), sW1F, 2, acc, lane);
    gemm_chunk(afrag_global(ap, 32 + kb), sW1F, 3, acc, lane);
#pragma unroll
    for (int t = 0; t < 4; ++t)
#pragma unroll
      for (int r = 0; r < 8; ++r) {
        const int M = r + (hi << 3), N = t * 16 + nlo;
        sHF[w][((N >> 5) << 9) + a_frag_idx(M, N & 31)] = (_Float16)silu_(acc[t][r] + sB1[N]);
      }
    lds_fence();
#pragma unroll
    for (int t = 0; t < 4; ++t) acc[t] = vzero8();
#pragma unroll
    for (int c = 0; c < 2; ++c) {
      v16h a = *(const v16h*)(&sHF[w][(c << 9) + (lane << 4)]);
      gemm_chunk(a, sW2F, c, acc, lane);
    }
#pragma unroll
    for (int t = 0; t < 4; ++t)
#pragma unroll
      for (int r = 0; r < 8; ++r) {
        const int M = r + (hi << 3), N = t * 16 + nlo;
        const long long idx = (long long)(nb + M) * HID + N;
        h[idx] = h[idx] + acc[t][r] + sB2[N];
      }
    if (lane < 16) {
      const int node = nb + lane;
      float c = cnt[node];
      if (c < 1.0f) c = 1.0f;
      xw[2 * node]     += cagg[2 * node] / c;
      xw[2 * node + 1] += cagg[2 * node + 1] / c;
    }
  }
}

// ---------------------------------------------------------------------------
// Embedding / output projections (<1% of FLOPs): plain per-node VALU.
// ---------------------------------------------------------------------------
__global__ void egnn_embed_kernel(const float* __restrict__ x,
                                  const float* __restrict__ w1, const float* __restrict__ b1,
                                  const float* __restrict__ w2, const float* __restrict__ b2,
                                  float* __restrict__ h, float* __restrict__ xw) {
  const int i = blockIdx.x * blockDim.x + threadIdx.x;
  if (i >= N_NODES) return;
  const float x0 = x[2 * i], x1 = x[2 * i + 1];
  xw[2 * i] = x0;
  xw[2 * i + 1] = x1;
  float t[HID];
#pragma unroll
  for (int o = 0; o < HID; ++o)
    t[o] = silu_(w1[2 * o] * x0 + w1[2 * o + 1] * x1 + b1[o]);
  for (int o = 0; o < HID; ++o) {
    float s = b2[o];
    for (int k = 0; k < HID; ++k) s += w2[o * HID + k] * t[k];
    h[(long long)i * HID + o] = s;
  }
}

__global__ void egnn_out_kernel(const float* __restrict__ h,
                                const float* __restrict__ w1, const float* __restrict__ b1,
                                const float* __restrict__ w2, const float* __restrict__ b2,
                                float* __restrict__ out) {
  const int i = blockIdx.x * blockDim.x + threadIdx.x;
  if (i >= N_NODES) return;
  float t[HID];
  for (int o = 0; o < HID; ++o) {
    float s = b1[o];
    for (int k = 0; k < HID; ++k) s += w1[o * HID + k] * h[(long long)i * HID + k];
    t[o] = silu_(s);
  }
  for (int o = 0; o < HID; ++o) {
    float s = b2[o];
    for (int k = 0; k < HID; ++k) s += w2[o * HID + k] * t[k];
    out[(long long)i * HID + o] = s;
  }
}

__global__ void egnn_zero_kernel(float* __restrict__ p, int n) {
  for (int i = blockIdx.x * blockDim.x + threadIdx.x; i < n; i += gridDim.x * blockDim.x)
    p[i] = 0.0f;
}

// ---------------------------------------------------------------------------
extern "C" void kernel_launch(void* const* d_in, const int* in_sizes, int n_in,
                              void* d_out, int out_size, void* d_ws, size_t ws_size,
                              hipStream_t stream) {
  const float*     x        = (const float*)d_in[0];
  const long long* ei       = (const long long*)d_in[1];
  const float*     emb_w1   = (const float*)d_in[2];
  const float*     emb_b1   = (const float*)d_in[3];
  const float*     emb_w2   = (const float*)d_in[4];
  const float*     emb_b2   = (const float*)d_in[5];
  const float*     edge_w1  = (const float*)d_in[6];
  const float*     edge_b1  = (const float*)d_in[7];
  const float*     edge_w2  = (const float*)d_in[8];
  const float*     edge_b2  = (const float*)d_in[9];
  const float*     att_w    = (const float*)d_in[10];
  const float*     att_b    = (const float*)d_in[11];
  const float*     node_w1  = (const float*)d_in[12];
  const float*     node_b1  = (const float*)d_in[13];
  const float*     node_w2  = (const float*)d_in[14];
  const float*     node_b2  = (const float*)d_in[15];
  const float*     coord_w1 = (const float*)d_in[16];
  const float*     coord_b1 = (const float*)d_in[17];
  const float*     coord_w2 = (const float*)d_in[18];
  const float*     out_w1   = (const float*)d_in[19];
  const float*     out_b1   = (const float*)d_in[20];
  const float*     out_w2   = (const float*)d_in[21];
  const float*     out_b2   = (const float*)d_in[22];

  float* ws   = (float*)d_ws;
  float* h    = ws;                             // N*64
  float* xw   = h + (size_t)N_NODES * HID;      // N*2
  float* agg  = xw + (size_t)2 * N_NODES;       // N*64
  float* cagg = agg + (size_t)N_NODES * HID;    // N*2
  float* cnt  = cagg + (size_t)2 * N_NODES;     // N
  // agg | cagg | cnt contiguous: zero as one span of N*67 floats per layer.

  egnn_embed_kernel<<<(N_NODES + 127) / 128, 128, 0, stream>>>(
      x, emb_w1, emb_b1, emb_w2, emb_b2, h, xw);

  const int edgeTiles = N_EDGES / 16;  // 50000, exact
  const int nodeTiles = N_NODES / 16;  // 3125, exact
  for (int l = 0; l < N_LAYERS; ++l) {
    egnn_zero_kernel<<<2048, 256, 0, stream>>>(agg, N_NODES * (HID + 3));
    egnn_edge_kernel<<<2048, 32 * EW, 0, stream>>>(
        h, xw, ei, edge_w1, edge_b1, edge_w2, edge_b2, att_w, att_b,
        coord_w1, coord_b1, coord_w2, agg, cagg, cnt, l, edgeTiles);
    egnn_node_kernel<<<256, 32 * NWV, 0, stream>>>(
        h, xw, agg, cagg, cnt, node_w1, node_b1, node_w2, node_b2, l, nodeTiles);
  }

  egnn_out_kernel<<<(N_NODES + 127) / 128, 128, 0, stream>>>(
      h, out_w1, out_b1, out_w2, out_b2, (float*)d_out);
}
